// EfficientSelfAttention_60035052863522
// MI455X (gfx1250) — matrix operand order repaired and checked by reading, not compiled, not run on hardware
//
#include <hip/hip_runtime.h>
#include <hip/hip_bf16.h>
#include <math.h>

// ---------- types / helpers ----------
typedef __bf16 bf16_t;
typedef bf16_t v16bf __attribute__((ext_vector_type(16)));
typedef float  v8f   __attribute__((ext_vector_type(8)));

union BFrag {
  v16bf v;
  uint4 u[2];
};

__device__ __forceinline__ unsigned short f2bf(float f) {
  union { float f; unsigned int u; } c; c.f = f;
  unsigned int u = c.u;
  u = (u + 0x7FFFu + ((u >> 16) & 1u)) >> 16;   // round-to-nearest-even
  return (unsigned short)u;
}

__device__ __forceinline__ unsigned int pack2bf(float lo, float hi) {
  return (unsigned int)f2bf(lo) | ((unsigned int)f2bf(hi) << 16);
}

__device__ __forceinline__ v8f vzero8() {
  v8f z;
#pragma unroll
  for (int i = 0; i < 8; ++i) z[i] = 0.0f;
  return z;
}

__device__ __forceinline__ v8f wmma_bf16(const BFrag& a, const BFrag& b, v8f c) {
  // (neg_a, A, neg_b, B, c_mod, C, reuse_a, reuse_b)
  return __builtin_amdgcn_wmma_f32_16x16x32_bf16(false, a.v, false, b.v,
                                                 (short)0, c, false, false);
}

#define HW     256
#define PLANE  65536     // 256*256
#define CIN    256
#define CQ     32
#define NROW   320       // 32 (Q) + 32 (K) + 256 (V)
#define L_WIN  4096
#define WSZ    64

// ---------- kernel 1: pack weights to bf16 ----------
__global__ __launch_bounds__(256) void prep_kernel(
    const float* __restrict__ Wq, const float* __restrict__ bq,
    const float* __restrict__ Wk, const float* __restrict__ bk,
    const float* __restrict__ Wv, const float* __restrict__ bv,
    unsigned short* __restrict__ Wb, float* __restrict__ ball)
{
  int i = blockIdx.x * 256 + threadIdx.x;
  if (i < NROW * CIN) {
    int row = i >> 8, col = i & 255;
    float v = (row < 32) ? Wq[row * CIN + col]
            : (row < 64) ? Wk[(row - 32) * CIN + col]
                         : Wv[(row - 64) * CIN + col];
    Wb[i] = f2bf(v);
  }
  if (i < NROW)
    ball[i] = (i < 32) ? bq[i] : (i < 64) ? bk[i - 32] : bv[i - 64];
}

// ---------- kernel 2: QKV 1x1-conv GEMM, Y[320x64] = W[320x256] * X[256x64] ----------
// outputs (layouts chosen for copy-through LDS staging in attn_kernel):
//   Qb[n][l][c]  bf16, Kb[n][l][c] bf16, Vb[n][ch][l] bf16
__global__ __launch_bounds__(256) void qkv_kernel(
    const float* __restrict__ x, const unsigned short* __restrict__ Wb,
    const float* __restrict__ ball,
    unsigned short* __restrict__ Qb, unsigned short* __restrict__ Kb,
    unsigned short* __restrict__ Vb)
{
  extern __shared__ char smem[];
  unsigned short* Wl = (unsigned short*)smem;                   // [320][32] bf16 k-slice
  unsigned short* Xt = (unsigned short*)(smem + NROW * 32 * 2); // [64][32] bf16 (X^T)
  float* Ys = (float*)smem;                                     // [320][64] f32 (reused)

  const int tid = threadIdx.x;
  const int wv  = tid >> 5;
  const int lane = tid & 31;
  const int lnN = lane & 15;
  const int g   = lane >> 4;
  const int n    = blockIdx.y;   // window
  const int tile = blockIdx.x;   // 64-token tile == window row ly
  const int wy = n >> 2, wx = n & 3;
  const int pixbase = (wy * WSZ + tile) * HW + wx * WSZ;

  v8f acc[10];
#pragma unroll
  for (int f = 0; f < 10; ++f) acc[f] = vzero8();

  for (int k0 = 0; k0 < CIN; k0 += 32) {
    __syncthreads();
    // weight slice: 320 rows x 32 cols = 1280 uint4
    for (int i = tid; i < NROW * 4; i += 256) {
      int row = i >> 2, part = i & 3;
      ((uint4*)Wl)[i] = *(const uint4*)(Wb + row * CIN + k0 + part * 8);
    }
    // x slice: 32 input channels x 64 pixels, f32 -> bf16, transposed
    for (int i = tid; i < 32 * 64; i += 256) {
      int cl = i >> 6, l = i & 63;
      float xv = x[(k0 + cl) * PLANE + pixbase + l];
      Xt[l * 32 + cl] = f2bf(xv);
    }
    __syncthreads();
    // B fragments (halves 0..7 -> K = 8g..8g+7 ; halves 8..15 -> K = 16+8g..23+8g), one per 16-col subtile
    BFrag B[4];
#pragma unroll
    for (int ci = 0; ci < 4; ++ci) {
      const uint4* p = (const uint4*)(Xt + (ci * 16 + lnN) * 32 + g * 8);
      B[ci].u[0] = p[0]; B[ci].u[1] = p[2];
    }
#pragma unroll
    for (int f = 0; f < 10; ++f) {
      int gf = wv * 10 + f;
      int ri = gf >> 2, ci = gf & 3;
      BFrag A;  // A: halves 0..7 -> K = 8g..8g+7 ; halves 8..15 -> K = 16+8g..23+8g
      const unsigned short* wrow = Wl + (ri * 16 + lnN) * 32;
      A.u[0] = *(const uint4*)(wrow + 8 * g);
      A.u[1] = *(const uint4*)(wrow + 16 + 8 * g);
      acc[f] = wmma_bf16(A, B[ci], acc[f]);
    }
  }
  __syncthreads();
#pragma unroll
  for (int f = 0; f < 10; ++f) {
    int gf = wv * 10 + f;
    int ri = gf >> 2, ci = gf & 3;
#pragma unroll
    for (int r = 0; r < 8; ++r)
      Ys[(ri * 16 + 8 * g + r) * 64 + ci * 16 + lnN] = acc[f][r];
  }
  __syncthreads();

  const long lbase = (long)n * L_WIN + (long)tile * 64;
  // Q rows 0..31 -> [n][l][c]
  for (int i = tid; i < 32 * 64; i += 256) {
    int l = i >> 5, c = i & 31;
    Qb[(lbase + l) * CQ + c] = f2bf(Ys[c * 64 + l] + ball[c]);
  }
  // K rows 32..63 -> [n][l][c]  (same layout as Q -> copy-through tile)
  for (int i = tid; i < 32 * 64; i += 256) {
    int l = i >> 5, c = i & 31;
    Kb[(lbase + l) * CQ + c] = f2bf(Ys[(32 + c) * 64 + l] + ball[32 + c]);
  }
  // V rows 64..319 -> [n][ch][l] (channel-major -> copy-through V^T tile)
  for (int i = tid; i < 256 * 64; i += 256) {
    int ch = i >> 6, l = i & 63;
    Vb[((long)n * CIN + ch) * L_WIN + tile * 64 + l] =
        f2bf(Ys[(64 + ch) * 64 + l] + ball[64 + ch]);
  }
}

// ---------- kernel 3: flash attention, 64-query tile / WG, double-buffered K/V ----------
__global__ __launch_bounds__(256) void attn_kernel(
    const float* __restrict__ x,
    const unsigned short* __restrict__ Qb, const unsigned short* __restrict__ Kb,
    const unsigned short* __restrict__ Vb, const float* __restrict__ gamma,
    float* __restrict__ out)
{
  extern __shared__ char smem[];
  unsigned short* Qs = (unsigned short*)smem;                    // [64][32]
  unsigned short* Kst0 = (unsigned short*)(smem + 4096);         // [64][32]
  unsigned short* Kst1 = (unsigned short*)(smem + 8192);         // [64][32]
  unsigned short* Vst0 = (unsigned short*)(smem + 12288);        // [256][64]
  unsigned short* Vst1 = (unsigned short*)(smem + 45056);        // [256][64]
  float*          Ss   = (float*)(smem + 77824);                 // [64][64]
  unsigned short* Ps   = (unsigned short*)(smem + 77824);        // overlays Ss
  float* red_max = (float*)(smem + 94208);                       // [4][64]
  float* red_sum = (float*)(smem + 95232);                       // [4][64]
  float* al_s    = (float*)(smem + 96256);                       // [64]
  float* l_s     = (float*)(smem + 96512);                       // [64]
  float* Os      = (float*)smem;                                 // [64][256] epilogue

  const int tid  = threadIdx.x;
  const int wv   = tid >> 5;
  const int lane = tid & 31;
  const int lnN  = lane & 15;
  const int g    = lane >> 4;
  const int n     = blockIdx.y;
  const int qtile = blockIdx.x;
  const long nL = (long)n * L_WIN;
  const long q0 = nL + (long)qtile * 64;
  const long vbase = (long)n * CIN * L_WIN;

  // Q tile: contiguous 4KB copy
  ((uint4*)Qs)[tid] = ((const uint4*)(Qb + q0 * CQ))[tid];

  v8f oac[4][2];
#pragma unroll
  for (int qi = 0; qi < 4; ++qi) { oac[qi][0] = vzero8(); oac[qi][1] = vzero8(); }
  float m_reg = -INFINITY;   // running row max (replicated x4 per row, consistent)
  float l_reg = 0.0f;        // running row sum (replicated x4 per row, consistent)

  const int row = tid & 63, qseg = tid >> 6;

  // preload tile 0 (copy-through: K tile is one contiguous 4KB block; V tile is
  // 128B-per-channel-row b128 bursts)
  ((uint4*)Kst0)[tid] = ((const uint4*)(Kb + nL * CQ))[tid];
#pragma unroll
  for (int p = 0; p < 8; ++p) {
    int i = p * 256 + tid, ch = i >> 3, part = i & 7;
    ((uint4*)(Vst0 + ch * 64))[part] =
        *(const uint4*)(Vb + vbase + (long)ch * L_WIN + part * 8);
  }

  for (int kt = 0; kt < 64; ++kt) {
    const int cur = kt & 1;
    const int k0 = kt * 64;
    unsigned short* Kc = cur ? Kst1 : Kst0;
    unsigned short* Vc = cur ? Vst1 : Vst0;
    unsigned short* Kn = cur ? Kst0 : Kst1;
    unsigned short* Vn = cur ? Vst0 : Vst1;
    __syncthreads();   // buf[cur] stores visible; buf[next] reads finished

    // software pipeline: issue next-tile global loads into registers now,
    // commit to LDS after the softmax (hides load latency behind S + softmax)
    uint4 kreg; uint4 vreg[8];
    const bool have_next = (kt + 1 < 64);
    if (have_next) {
      kreg = ((const uint4*)(Kb + (nL + k0 + 64) * CQ))[tid];
#pragma unroll
      for (int p = 0; p < 8; ++p) {
        int i = p * 256 + tid, ch = i >> 3, part = i & 7;
        vreg[p] = *(const uint4*)(Vb + vbase + (long)ch * L_WIN + k0 + 64 + part * 8);
      }
      if (kt + 2 < 64)  // gfx1250 global_prefetch_b8 two tiles ahead
        __builtin_prefetch(Vb + vbase + (long)tid * L_WIN + k0 + 128, 0, 1);
    }

    // ---- S = Q^T K : 16 subtiles, 2 per wave (K-dim = Cq = 32 = one WMMA) ----
#pragma unroll
    for (int t = 0; t < 2; ++t) {
      int st = wv * 2 + t;
      int qi = st >> 2, ki = st & 3;
      BFrag A, Bm;
      const unsigned short* qrow = Qs + (qi * 16 + lnN) * 32;
      A.u[0] = *(const uint4*)(qrow + 8 * g);
      A.u[1] = *(const uint4*)(qrow + 16 + 8 * g);
      const unsigned short* krow = Kc + (ki * 16 + lnN) * 32;
      Bm.u[0] = *(const uint4*)(krow + 8 * g);
      Bm.u[1] = *(const uint4*)(krow + 16 + 8 * g);
      v8f s = wmma_bf16(A, Bm, vzero8());
#pragma unroll
      for (int r = 0; r < 8; ++r)
        Ss[(qi * 16 + 8 * g + r) * 64 + ki * 16 + lnN] = s[r];
    }
    __syncthreads();

    // ---- online softmax, stage A: vectorized score reads + partial max ----
    float seg[16];
    {
      const float4* srow = (const float4*)(Ss + row * 64 + qseg * 16);
#pragma unroll
      for (int q4 = 0; q4 < 4; ++q4) {
        float4 v = srow[q4];
        seg[q4 * 4 + 0] = v.x; seg[q4 * 4 + 1] = v.y;
        seg[q4 * 4 + 2] = v.z; seg[q4 * 4 + 3] = v.w;
      }
    }
    float pm = seg[0];
#pragma unroll
    for (int j = 1; j < 16; ++j) pm = fmaxf(pm, seg[j]);
    red_max[qseg * 64 + row] = pm;
    __syncthreads();

    // ---- stage B (all threads, per-row replicated): max/alpha/exp/pack ----
    float tmax = fmaxf(fmaxf(red_max[row], red_max[64 + row]),
                       fmaxf(red_max[128 + row], red_max[192 + row]));
    float mn    = fmaxf(m_reg, tmax);
    float alpha = __expf(m_reg - mn);
    m_reg = mn;
    if (qseg == 0) al_s[row] = alpha;
    float psum = 0.0f;
    float e[16];
#pragma unroll
    for (int j = 0; j < 16; ++j) { e[j] = __expf(seg[j] - mn); psum += e[j]; }
    {
      uint4 pk0, pk1;   // 16 bf16 packed probabilities
      pk0.x = pack2bf(e[0], e[1]);   pk0.y = pack2bf(e[2], e[3]);
      pk0.z = pack2bf(e[4], e[5]);   pk0.w = pack2bf(e[6], e[7]);
      pk1.x = pack2bf(e[8], e[9]);   pk1.y = pack2bf(e[10], e[11]);
      pk1.z = pack2bf(e[12], e[13]); pk1.w = pack2bf(e[14], e[15]);
      uint4* prow = (uint4*)(Ps + row * 64 + qseg * 16);  // seg[] already in regs
      prow[0] = pk0; prow[1] = pk1;
    }
    red_sum[qseg * 64 + row] = psum;
    __syncthreads();

    // ---- stage C: finish running sum, commit next tile, rescale, O-GEMM ----
    l_reg = l_reg * alpha + red_sum[row] + red_sum[64 + row]
                          + red_sum[128 + row] + red_sum[192 + row];
    if (have_next) {
      ((uint4*)Kn)[tid] = kreg;
#pragma unroll
      for (int p = 0; p < 8; ++p) {
        int i = p * 256 + tid, ch = i >> 3, part = i & 7;
        ((uint4*)(Vn + ch * 64))[part] = vreg[p];
      }
    }
#pragma unroll
    for (int qi = 0; qi < 4; ++qi)
#pragma unroll
      for (int r = 0; r < 8; ++r) {
        float al = al_s[qi * 16 + 8 * g + r];
        oac[qi][0][r] *= al;
        oac[qi][1][r] *= al;
      }
    // O += P * V^T  (each wave owns 32 channels; 16 WMMAs)
#pragma unroll
    for (int kk = 0; kk < 2; ++kk) {
      BFrag Bv0, Bv1;
      {
        int ch0 = wv * 32 + lnN;
        const unsigned short* v0 = Vc + ch0 * 64 + kk * 32 + 8 * g;
        Bv0.u[0] = *(const uint4*)v0; Bv0.u[1] = *(const uint4*)(v0 + 16);
        const unsigned short* v1 = Vc + (ch0 + 16) * 64 + kk * 32 + 8 * g;
        Bv1.u[0] = *(const uint4*)v1; Bv1.u[1] = *(const uint4*)(v1 + 16);
      }
#pragma unroll
      for (int qi = 0; qi < 4; ++qi) {
        BFrag A;
        const unsigned short* prow = Ps + (qi * 16 + lnN) * 64 + kk * 32;
        A.u[0] = *(const uint4*)(prow + 8 * g);
        A.u[1] = *(const uint4*)(prow + 16 + 8 * g);
        oac[qi][0] = wmma_bf16(A, Bv0, oac[qi][0]);
        oac[qi][1] = wmma_bf16(A, Bv1, oac[qi][1]);
      }
    }
  }

  // publish row sums once, then normalize in fragment layout
  if (qseg == 0) l_s[row] = l_reg;
  __syncthreads();
#pragma unroll
  for (int qi = 0; qi < 4; ++qi)
#pragma unroll
    for (int r = 0; r < 8; ++r) {
      float inv = 1.0f / l_s[qi * 16 + 8 * g + r];
      oac[qi][0][r] *= inv;
      oac[qi][1][r] *= inv;
    }
  __syncthreads();
  // stage O in LDS with +row column swizzle (conflict-free both directions)
#pragma unroll
  for (int qi = 0; qi < 4; ++qi)
#pragma unroll
    for (int cj = 0; cj < 2; ++cj) {
      int ch = wv * 32 + cj * 16 + lnN;
#pragma unroll
      for (int r = 0; r < 8; ++r) {
        int qr = qi * 16 + 8 * g + r;
        Os[qr * 256 + ((ch + qr) & 255)] = oac[qi][cj][r];
      }
    }
  __syncthreads();

  const float gam = gamma[0];
  const int wy = n >> 2, wx = n & 3;
  const int pixbase = (wy * WSZ + qtile) * HW + wx * WSZ;
  for (int i = tid; i < 256 * 64; i += 256) {
    int ch = i >> 6, lx = i & 63;
    int gidx = ch * PLANE + pixbase + lx;
    float o = Os[lx * 256 + ((ch + lx) & 255)];
    out[gidx] = gam * o + x[gidx];
  }
}

// ---------- launch ----------
extern "C" void kernel_launch(void* const* d_in, const int* in_sizes, int n_in,
                              void* d_out, int out_size, void* d_ws, size_t ws_size,
                              hipStream_t stream) {
  (void)in_sizes; (void)n_in; (void)out_size; (void)ws_size;
  const float* x  = (const float*)d_in[0];
  const float* Wq = (const float*)d_in[1];
  const float* bq = (const float*)d_in[2];
  const float* Wk = (const float*)d_in[3];
  const float* bk = (const float*)d_in[4];
  const float* Wv = (const float*)d_in[5];
  const float* bv = (const float*)d_in[6];
  const float* gm = (const float*)d_in[7];
  float* out = (float*)d_out;

  char* ws = (char*)d_ws;
  unsigned short* Wb   = (unsigned short*)ws;                       // 320*256*2     = 163840
  float*          ball = (float*)(ws + 163840);                     // 320*4         = 1280
  unsigned short* Qb   = (unsigned short*)(ws + 165120);            // 16*4096*32*2  = 4 MiB
  unsigned short* Kb   = (unsigned short*)(ws + 165120 + 4194304);  // 4 MiB
  unsigned short* Vb   = (unsigned short*)(ws + 165120 + 8388608);  // 16*256*4096*2 = 32 MiB

  prep_kernel<<<320, 256, 0, stream>>>(Wq, bq, Wk, bk, Wv, bv, Wb, ball);
  qkv_kernel<<<dim3(64, 16), 256, 81920, stream>>>(x, Wb, ball, Qb, Kb, Vb);
  attn_kernel<<<dim3(64, 16), 256, 96768, stream>>>(x, Qb, Kb, Vb, gm, out);
}
